// EmbLoss_v1_44452911514024
// MI455X (gfx1250) — matrix operand order repaired and checked by reading, not compiled, not run on hardware
//
#include <hip/hip_runtime.h>
#include <hip/hip_bf16.h>

typedef __attribute__((ext_vector_type(16))) _Float16 v16h;
typedef __attribute__((ext_vector_type(8)))  float    v8f;

#define IMG_H 640
#define IMG_W 640
#define P_IMG (IMG_H * IMG_W)      // 409600 pixels per image
#define N_IMG 16
#define N_LAB 8
#define BLOCKS_PER_IMG_1 50        // 50 blocks * 8 waves * 32 iters * 32 lanes = 409600
#define BLOCKS_PER_IMG_2 64        // 64 blocks * 256 thr * 25 iters = 409600

// ws layout (floats): ws0 = [16 images][8 labels][8]: cols 0..3 = emb sums, col 4 = count
//                     ws1 = [16 images][16]: [0..7] = sum of agg terms, [8..15] = inst counts
#define WS0_FLOATS (N_IMG * 64)
#define WS1_FLOATS (N_IMG * 16)

__global__ void zero_ws_kernel(float* ws) {
    for (int i = threadIdx.x; i < WS0_FLOATS + WS1_FLOATS; i += blockDim.x) ws[i] = 0.0f;
}

// ---------------------------------------------------------------------------
// Pass 1: kernel-region segment sums via WMMA.
//   sums[l][c] = sum_p onehot(inst_k[p]==l) * emb[p][c],  counts via B column 4 = 1.0
//   f32 fidelity: emb = hi(f16) + lo(f16 residual); D = A*Bhi + A*Blo (A one-hot, exact)
// ---------------------------------------------------------------------------
__global__ __launch_bounds__(256) void pass1_segsum_wmma(
    const float4* __restrict__ emb4, const int* __restrict__ inst,
    const float* __restrict__ ker, const float* __restrict__ tmask,
    float* __restrict__ ws0)
{
    __shared__ unsigned  sInst[8][32];
    __shared__ _Float16  sHi[8][32][4];
    __shared__ _Float16  sLo[8][32][4];

    const int w    = threadIdx.x >> 5;
    const int lane = threadIdx.x & 31;
    const int b    = blockIdx.x / BLOCKS_PER_IMG_1;
    const int cib  = blockIdx.x % BLOCKS_PER_IMG_1;
    const size_t imgBase = (size_t)b * P_IMG;
    const int waveBase = (cib * 8 + w) * 1024;   // 1024 contiguous pixels per wave

    const int h = lane >> 4;        // lane half-group
    const int m = lane & 15;        // A: label row   / B,D: column index

    v8f acc = {};                   // 16x16 f32 accumulator (labels x [ch0..3,count,...])

    for (int it = 0; it < 32; ++it) {
        const size_t gi = imgBase + (size_t)(waveBase + it * 32 + lane);

        // coalesced 128-bit + 32-bit loads
        const float4 e  = emb4[gi];
        const int    iv = inst[gi];
        const float  kv = ker[gi];
        const float  tv = tmask[gi];
        const int instK = (kv > 0.5f && tv > 0.5f) ? iv : 0;

        const _Float16 h0 = (_Float16)e.x, h1 = (_Float16)e.y,
                       h2 = (_Float16)e.z, h3 = (_Float16)e.w;

        sInst[w][lane]  = (unsigned)instK;
        sHi[w][lane][0] = h0; sHi[w][lane][1] = h1;
        sHi[w][lane][2] = h2; sHi[w][lane][3] = h3;
        sLo[w][lane][0] = (_Float16)(e.x - (float)h0);
        sLo[w][lane][1] = (_Float16)(e.y - (float)h1);
        sLo[w][lane][2] = (_Float16)(e.z - (float)h2);
        sLo[w][lane][3] = (_Float16)(e.w - (float)h3);

        // wave-private LDS region: only need store->load ordering within this wave
        asm volatile("s_wait_dscnt 0" ::: "memory");

        // A: 16x32 f16 one-hot. half j of lane -> K = 8h+j (j<8) else 16+8h+(j-8)
        v16h A;
        #pragma unroll
        for (int j = 0; j < 16; ++j) {
            const int k = (j < 8) ? (8 * h + j) : (16 + 8 * h + (j - 8));
            A[j] = (sInst[w][k] == (unsigned)m) ? (_Float16)1.0f : (_Float16)0.0f;
        }

        // B: 32x16 f16. half j of lane -> K = 8h+j (j<8) else 16+8h+(j-8), column n = lane&15
        v16h Bh, Bl;
        if (m < 4) {
            #pragma unroll
            for (int j = 0; j < 16; ++j) {
                const int k = (j < 8) ? (8 * h + j) : (16 + 8 * h + (j - 8));
                Bh[j] = sHi[w][k][m];
                Bl[j] = sLo[w][k][m];
            }
        } else {
            const _Float16 one = (m == 4) ? (_Float16)1.0f : (_Float16)0.0f;
            #pragma unroll
            for (int j = 0; j < 16; ++j) { Bh[j] = one; Bl[j] = (_Float16)0.0f; }
        }

        acc = __builtin_amdgcn_wmma_f32_16x16x32_f16(false, A, false, Bh,
                                                     (short)0, acc, false, false);
        acc = __builtin_amdgcn_wmma_f32_16x16x32_f16(false, A, false, Bl,
                                                     (short)0, acc, false, false);
    }

    // D layout: lanes 0..15 -> N=lane, VGPR r -> label M=r (labels 8..15 on lanes 16..31, all zero)
    if (lane < 5) {
        #pragma unroll
        for (int r = 0; r < N_LAB; ++r)
            atomicAdd(&ws0[b * 64 + r * 8 + lane], acc[r]);
    }
}

// ---------------------------------------------------------------------------
// Pass 2: per-pixel aggregation term  log(max(||e - mu[inst]|| - 0.5, 0)^2 + 1)
// segment-summed per label + per-label pixel counts.
// ---------------------------------------------------------------------------
__global__ __launch_bounds__(256) void pass2_agg(
    const float4* __restrict__ emb4, const int* __restrict__ inst,
    const float* __restrict__ tmask,
    const float* __restrict__ ws0, float* __restrict__ ws1)
{
    __shared__ float sMu[N_LAB][4];

    const int b   = blockIdx.x / BLOCKS_PER_IMG_2;
    const int blk = blockIdx.x % BLOCKS_PER_IMG_2;

    if (threadIdx.x < N_LAB) {
        const int l = threadIdx.x;
        const float inv = 1.0f / fmaxf(ws0[b * 64 + l * 8 + 4], 1.0f);
        #pragma unroll
        for (int c = 0; c < 4; ++c)
            sMu[l][c] = (l == 0) ? 0.0f : ws0[b * 64 + l * 8 + c] * inv;
    }
    __syncthreads();

    const size_t imgBase = (size_t)b * P_IMG;
    const int base = blk * (P_IMG / BLOCKS_PER_IMG_2);

    float sumT[N_LAB], sumC[N_LAB];
    #pragma unroll
    for (int l = 0; l < N_LAB; ++l) { sumT[l] = 0.0f; sumC[l] = 0.0f; }

    for (int i = 0; i < 25; ++i) {
        const size_t gi = imgBase + (size_t)(base + i * 256 + threadIdx.x);
        const float4 e = emb4[gi];
        const int iv = inst[gi];
        const int ie = (tmask[gi] > 0.5f) ? iv : 0;

        const float4 mv = *(const float4*)sMu[ie];
        const float dx = e.x - mv.x, dy = e.y - mv.y, dz = e.z - mv.z, dw = e.w - mv.w;
        const float sq = dx * dx + dy * dy + dz * dz + dw * dw;
        const float d  = (sq > 0.0f) ? __fsqrt_rn(sq) : 0.0f;
        const float r  = fmaxf(d - 0.5f, 0.0f);
        const float t  = __logf(r * r + 1.0f);

        #pragma unroll
        for (int l = 0; l < N_LAB; ++l) {
            const bool eq = (ie == l);
            sumT[l] += eq ? t : 0.0f;
            sumC[l] += eq ? 1.0f : 0.0f;
        }
    }

    const int lane = threadIdx.x & 31;
    #pragma unroll
    for (int l = 0; l < N_LAB; ++l) {
        float s = sumT[l], c = sumC[l];
        #pragma unroll
        for (int off = 16; off > 0; off >>= 1) {
            s += __shfl_xor(s, off);
            c += __shfl_xor(c, off);
        }
        if (lane == 0) {
            atomicAdd(&ws1[b * 16 + l], s);
            atomicAdd(&ws1[b * 16 + 8 + l], c);
        }
    }
}

// ---------------------------------------------------------------------------
// Finalize: l_agg = mean(agg[2:]); l_dis over ordered pairs of labels 1..7;
// l_reg; batch mean.
// ---------------------------------------------------------------------------
__global__ void finalize_kernel(const float* __restrict__ ws0,
                                const float* __restrict__ ws1,
                                float* __restrict__ out)
{
    __shared__ float red[N_IMG];
    const int t = threadIdx.x;
    if (t < N_IMG) {
        const int b = t;
        float mu[N_LAB][4];
        #pragma unroll
        for (int l = 0; l < N_LAB; ++l) {
            const float inv = 1.0f / fmaxf(ws0[b * 64 + l * 8 + 4], 1.0f);
            #pragma unroll
            for (int c = 0; c < 4; ++c)
                mu[l][c] = (l == 0) ? 0.0f : ws0[b * 64 + l * 8 + c] * inv;
        }
        // aggregation: reference drops labels 0 and 1 (mean of agg[2:])
        float l_agg = 0.0f;
        #pragma unroll
        for (int l = 2; l < N_LAB; ++l)
            l_agg += ws1[b * 16 + l] / fmaxf(ws1[b * 16 + 8 + l], 1.0f);
        l_agg *= (1.0f / 6.0f);
        // discrimination: 7*6 = 42 ordered pairs among labels 1..7
        float l_dis = 0.0f;
        for (int l1 = 1; l1 < N_LAB; ++l1)
            for (int l2 = 1; l2 < N_LAB; ++l2) {
                if (l1 == l2) continue;
                float sq = 0.0f;
                #pragma unroll
                for (int c = 0; c < 4; ++c) {
                    const float d = mu[l1][c] - mu[l2][c];
                    sq += d * d;
                }
                const float dd = (sq > 0.0f) ? __fsqrt_rn(sq) : 0.0f;
                const float r  = fmaxf(3.0f - dd, 0.0f);
                l_dis += __logf(r * r + 1.0f);
            }
        l_dis *= (1.0f / 42.0f);
        // regularizer over all 8 rows (row 0 contributes log(1)=0)
        float l_reg = 0.0f;
        #pragma unroll
        for (int l = 0; l < N_LAB; ++l) {
            float sq = 0.0f;
            #pragma unroll
            for (int c = 0; c < 4; ++c) sq += mu[l][c] * mu[l][c];
            const float nrm = (sq > 0.0f) ? __fsqrt_rn(sq) : 0.0f;
            l_reg += __logf(nrm + 1.0f);
        }
        l_reg *= (0.001f / 8.0f);
        red[b] = l_agg + l_dis + l_reg;
    }
    __syncthreads();
    if (t == 0) {
        float s = 0.0f;
        for (int i = 0; i < N_IMG; ++i) s += red[i];
        out[0] = s * (1.0f / (float)N_IMG);   // LOSS_WEIGHT = 1.0
    }
}

extern "C" void kernel_launch(void* const* d_in, const int* in_sizes, int n_in,
                              void* d_out, int out_size, void* d_ws, size_t ws_size,
                              hipStream_t stream) {
    const float4* emb4  = (const float4*)d_in[0];
    const int*    inst  = (const int*)d_in[1];
    const float*  ker   = (const float*)d_in[2];
    const float*  tmask = (const float*)d_in[3];
    float* ws0 = (float*)d_ws;
    float* ws1 = ws0 + WS0_FLOATS;
    float* out = (float*)d_out;

    zero_ws_kernel<<<1, 256, 0, stream>>>(ws0);
    pass1_segsum_wmma<<<dim3(N_IMG * BLOCKS_PER_IMG_1), 256, 0, stream>>>(
        emb4, inst, ker, tmask, ws0);
    pass2_agg<<<dim3(N_IMG * BLOCKS_PER_IMG_2), 256, 0, stream>>>(
        emb4, inst, tmask, ws0, ws1);
    finalize_kernel<<<1, 64, 0, stream>>>(ws0, ws1, out);
}
